// MLPModel_63075889709246
// MI455X (gfx1250) — matrix operand order repaired and checked by reading, not compiled, not run on hardware
//
#include <hip/hip_runtime.h>

// ---------------------------------------------------------------------------
// MI455X (gfx1250, wave32) implementation of: 3x (dense -> Newton-Schulz ZCA)
// All heavy GEMMs use v_wmma_f32_16x16x32_bf16 (bf16 operands, f32 accum),
// with LDS staging pre-swizzled into the WMMA fragment layout so fragment
// reads are ds_load_b128, and all four B fragments hoisted ahead of the WMMA
// chain so the four WMMAs issue back-to-back after a single s_wait_dscnt.
// Deterministic: no atomics; split-K partials reduced by dedicated kernels.
// ---------------------------------------------------------------------------

typedef __attribute__((ext_vector_type(16))) __bf16 v16bf;
typedef __attribute__((ext_vector_type(8)))  float  v8f;

union Frag16 { v16bf v; uint4 q[2]; unsigned short s[16]; };

__device__ __forceinline__ unsigned short f2bf(float f) {
  unsigned int u = __float_as_uint(f);
  u += 0x7FFFu + ((u >> 16) & 1u);          // round-to-nearest-even
  return (unsigned short)(u >> 16);
}

// ---------------------------------------------------------------------------
// GEMM: D[M,N] = alpha*(A_bf16 @ B_bf16) + beta*Cin  (row-major, f32 out)
//  - optional bf16 copy of D (for chained bf16 GEMMs)
//  - optional split-K: blockIdx.z processes K-chunk `kchunk`, writing to
//    D + blockIdx.z*dsplit (launch with gridDim.z=1, kchunk=K, dsplit=0 for
//    a normal GEMM).
// Block tile 128x64, 256 threads = 8 waves, wave tile 16x64 (4 wmma accums).
// ---------------------------------------------------------------------------
__global__ __launch_bounds__(256)
void gemm_bf16(const unsigned short* __restrict__ A, int lda,
               const unsigned short* __restrict__ B, int ldb,
               float* __restrict__ D, int ldd,
               unsigned short* __restrict__ Dbf,
               const float* __restrict__ Cin,
               float alpha, float beta,
               int kchunk, size_t dsplit)
{
  __shared__ uint4 sA4[512];                          // 128 rows x 32 bf16 (8 KB)
  __shared__ __align__(16) unsigned short sBsw[2048]; // swizzled 32x64 tile (4 KB)
  unsigned short* sA = (unsigned short*)sA4;

  const int tid  = threadIdx.x;
  const int wave = tid >> 5;
  const int lane = tid & 31;
  const int m0 = blockIdx.x * 128;
  const int n0 = blockIdx.y * 64;
  const int kbeg = blockIdx.z * kchunk;
  const int kend = kbeg + kchunk;

  const v8f zero = {0.f,0.f,0.f,0.f,0.f,0.f,0.f,0.f};
  v8f acc[4];
  #pragma unroll
  for (int j = 0; j < 4; ++j) acc[j] = zero;

  for (int k0 = kbeg; k0 < kend; k0 += 32) {
    // Stage A tile (128x32) as b128 loads: 512 uint4, 2 per thread.
    #pragma unroll
    for (int t = 0; t < 2; ++t) {
      int u = tid + t * 256;
      int row = u >> 2, q = u & 3;
      sA4[u] = ((const uint4*)(A + (size_t)(m0 + row) * lda + k0))[q];
    }
    // Stage B tile (32x64), scattering into the WMMA fragment layout:
    //   sBsw[j*512 + l*16 + h] = B[kb(l)+(h&7)+16*(h>>3)][n0 + j*16 + (l&15)], kb(l)=8*(l>>4)
    {
      int k  = tid >> 3;          // 0..31
      int ng = tid & 7;           // uint4 group along n
      uint4 v = ((const uint4*)(B + (size_t)(k0 + k) * ldb + n0))[ng];
      unsigned short e[8];
      *(uint4*)e = v;
      const int lhalf = (k & 8) ? 16 : 0;
      const int h = (k & 7) | ((k >> 4) << 3);
      #pragma unroll
      for (int x = 0; x < 8; ++x) {
        int n = ng * 8 + x;
        int j = n >> 4;
        int l = (n & 15) | lhalf;
        sBsw[j * 512 + l * 16 + h] = e[x];
      }
    }
    if (k0 + 32 < kend) {
      __builtin_prefetch(A + (size_t)(m0 + (tid & 127)) * lda + k0 + 32, 0, 1);
      if (tid < 32)
        __builtin_prefetch(B + (size_t)(k0 + 32 + tid) * ldb + n0, 0, 1);
    }
    __syncthreads();

    // A fragment (ISA 16-bit A 16x32 layout): lane m holds K {kb..kb+7, kb+16..kb+23}
    Frag16 fa;
    {
      const int m  = lane & 15;
      const int kb = (lane < 16) ? 0 : 8;
      const unsigned short* row = sA + (wave * 16 + m) * 32 + kb;
      fa.q[0] = *(const uint4*)(row);        // kb .. kb+7
      fa.q[1] = *(const uint4*)(row + 16);   // kb+16 .. kb+23
    }
    // Hoist all four B fragments (2x ds_load_b128 each) ahead of the WMMA
    // chain: one dscnt wait, then 4 independent back-to-back WMMAs.
    Frag16 fb[4];
    #pragma unroll
    for (int j = 0; j < 4; ++j) {
      const unsigned short* bp = sBsw + j * 512 + lane * 16;
      fb[j].q[0] = ((const uint4*)bp)[0];
      fb[j].q[1] = ((const uint4*)bp)[1];
    }
    #pragma unroll
    for (int j = 0; j < 4; ++j) {
      acc[j] = __builtin_amdgcn_wmma_f32_16x16x32_bf16(
                 false, fa.v, false, fb[j].v, (short)0, acc[j], false, false);
    }
    __syncthreads();
  }

  // Epilogue (ISA 32-bit C/D 16x16 layout): lanes0-15 rows r, lanes16-31 rows r+8
  float* __restrict__ Dp = D + (size_t)blockIdx.z * dsplit;
  const int n  = lane & 15;
  const int mb = (lane < 16) ? 0 : 8;
  #pragma unroll
  for (int j = 0; j < 4; ++j) {
    const int gm = m0 + wave * 16 + mb;
    const int gn = n0 + j * 16 + n;
    #pragma unroll
    for (int r = 0; r < 8; ++r) {
      float v = alpha * acc[j][r];
      if (Cin) v += beta * Cin[(size_t)(gm + r) * ldd + gn];
      Dp[(size_t)(gm + r) * ldd + gn] = v;
      if (Dbf) Dbf[(size_t)(gm + r) * ldd + gn] = f2bf(v);
    }
  }
}

// ---------------------------------------------------------------------------
// LDS-tiled bf16 transpose: dst[c][r] = src[r][c], src is [65536 x 512].
// 64x64 tiles; all global accesses are b128; LDS padded to keep b128 stores.
// ---------------------------------------------------------------------------
__global__ __launch_bounds__(256)
void transpose_bf16(const unsigned short* __restrict__ src,
                    unsigned short* __restrict__ dst)
{
  __shared__ __align__(16) unsigned short tile[64 * 72];
  const int r0 = blockIdx.x * 64;   // source row block (65536/64)
  const int c0 = blockIdx.y * 64;   // source col block (512/64)
  const int tid = threadIdx.x;

  #pragma unroll
  for (int t = 0; t < 2; ++t) {
    int u = tid + t * 256;
    int row = u >> 3, q = u & 7;
    *(uint4*)(tile + row * 72 + q * 8) =
        *(const uint4*)(src + (size_t)(r0 + row) * 512 + c0 + q * 8);
  }
  __syncthreads();
  #pragma unroll
  for (int t = 0; t < 2; ++t) {
    int u = tid + t * 256;
    int orow = u >> 3, q = u & 7;   // orow = source column
    unsigned short tmp[8];
    #pragma unroll
    for (int e = 0; e < 8; ++e) tmp[e] = tile[(q * 8 + e) * 72 + orow];
    *(uint4*)(dst + (size_t)(c0 + orow) * 65536 + r0 + q * 8) = *(uint4*)tmp;
  }
}

__global__ __launch_bounds__(256)
void covreduce_kernel(const float* __restrict__ covpart, float* __restrict__ covacc)
{
  int idx = blockIdx.x * 256 + threadIdx.x;   // 262144 total
  float s = 0.f;
  #pragma unroll
  for (int p = 0; p < 32; ++p) s += covpart[(size_t)p * 262144 + idx];
  covacc[idx] = s;
}

// ---------------------------------------------------------------------------
// Elementwise / reduction helpers
// ---------------------------------------------------------------------------
__global__ __launch_bounds__(256)
void cvt_kernel(const float* __restrict__ src, unsigned short* __restrict__ dst, int n)
{
  for (int i = blockIdx.x * 256 + threadIdx.x; i < n; i += gridDim.x * 256)
    dst[i] = f2bf(src[i]);
}

__global__ __launch_bounds__(256)
void cvt_scale_kernel(const float* __restrict__ src, const float* __restrict__ scale,
                      unsigned short* __restrict__ dst, int n)
{
  const float s = *scale;
  for (int i = blockIdx.x * 256 + threadIdx.x; i < n; i += gridDim.x * 256)
    dst[i] = f2bf(src[i] * s);
}

__global__ __launch_bounds__(256)
void colsum_kernel(const float* __restrict__ H, float* __restrict__ part)
{
  const int r0 = blockIdx.x * 256;
  const int c  = threadIdx.x;
  float a0 = 0.f, a1 = 0.f;
  for (int r = 0; r < 256; ++r) {
    const float* row = H + (size_t)(r0 + r) * 512;
    a0 += row[c];
    a1 += row[c + 256];
  }
  part[(size_t)blockIdx.x * 512 + c]       = a0;
  part[(size_t)blockIdx.x * 512 + c + 256] = a1;
}

__global__ __launch_bounds__(256)
void meanreduce_kernel(const float* __restrict__ part, float* __restrict__ meansum)
{
  const int c = blockIdx.x * 256 + threadIdx.x;   // 512 total
  float s = 0.f;
  for (int b = 0; b < 256; ++b) s += part[(size_t)b * 512 + c];
  meansum[c] = s;
}

__global__ __launch_bounds__(256)
void center_kernel(const float* __restrict__ H, const float* __restrict__ meansum,
                   unsigned short* __restrict__ Xcb, int total)
{
  const float inv = 1.0f / 65536.0f;
  for (int i = blockIdx.x * 256 + threadIdx.x; i < total; i += gridDim.x * 256) {
    const int c = i & 511;
    Xcb[i] = f2bf(H[i] - meansum[c] * inv);
  }
}

__global__ __launch_bounds__(256)
void trace_kernel(const float* __restrict__ covacc, float* __restrict__ scal)
{
  __shared__ float red[256];
  const int t = threadIdx.x;
  const float invn = 1.0f / 65536.0f;
  float s = covacc[(size_t)t * 513] * invn + 0.1f;            // + I/temp, temp=10
  s      += covacc[(size_t)(t + 256) * 513] * invn + 0.1f;
  red[t] = s;
  __syncthreads();
  for (int off = 128; off > 0; off >>= 1) {
    if (t < off) red[t] += red[t + off];
    __syncthreads();
  }
  if (t == 0) { scal[0] = red[0]; scal[1] = rsqrtf(red[0]); }
}

__global__ __launch_bounds__(256)
void sconvert_kernel(const float* __restrict__ covacc, const float* __restrict__ scal,
                     unsigned short* __restrict__ Sb)
{
  const int idx = blockIdx.x * 256 + threadIdx.x;   // 262144 total
  const int i = idx >> 9, j = idx & 511;
  const float invn = 1.0f / 65536.0f;
  float cov = covacc[idx] * invn + ((i == j) ? 0.1f : 0.f);
  Sb[idx] = f2bf(cov / scal[0]);
}

__global__ __launch_bounds__(256)
void pinit_kernel(float* __restrict__ P, unsigned short* __restrict__ Pb)
{
  const int idx = blockIdx.x * 256 + threadIdx.x;   // 262144 total
  const float v = ((idx >> 9) == (idx & 511)) ? 1.f : 0.f;
  P[idx]  = v;
  Pb[idx] = f2bf(v);
}

// ---------------------------------------------------------------------------
// Host driver
// ---------------------------------------------------------------------------
extern "C" void kernel_launch(void* const* d_in, const int* in_sizes, int n_in,
                              void* d_out, int out_size, void* d_ws, size_t ws_size,
                              hipStream_t stream)
{
  (void)in_sizes; (void)n_in; (void)out_size; (void)ws_size;
  const size_t ND = 65536ull * 512ull;   // 33,554,432
  const size_t DD = 512ull * 512ull;     // 262,144

  const float* x = (const float*)d_in[0];
  const float* W[3] = { (const float*)d_in[1], (const float*)d_in[2], (const float*)d_in[3] };
  float* out = (float*)d_out;

  char* p = (char*)d_ws;
  auto take = [&](size_t bytes) -> void* {
    void* r = (void*)p;
    p += (bytes + 255) & ~(size_t)255;
    return r;
  };
  float*          H       = (float*)take(ND * 4);
  unsigned short* Ab      = (unsigned short*)take(ND * 2);
  unsigned short* Xcb     = (unsigned short*)take(ND * 2);
  unsigned short* XcT     = (unsigned short*)take(ND * 2);   // [512 x 65536]
  unsigned short* Wb      = (unsigned short*)take(DD * 2);
  float*          covpart = (float*)take(32ull * DD * 4);
  float*          covacc  = (float*)take(DD * 4);
  unsigned short* Sb      = (unsigned short*)take(DD * 2);
  float*          P0      = (float*)take(DD * 4);
  float*          P1      = (float*)take(DD * 4);
  unsigned short* Pb0     = (unsigned short*)take(DD * 2);
  unsigned short* Pb1     = (unsigned short*)take(DD * 2);
  float*          T1      = (float*)take(DD * 4);
  unsigned short* T1b     = (unsigned short*)take(DD * 2);
  float*          T2      = (float*)take(DD * 4);
  unsigned short* T2b     = (unsigned short*)take(DD * 2);
  unsigned short* Cb      = (unsigned short*)take(DD * 2);
  float*          colpart = (float*)take(256ull * 512 * 4);
  float*          meansum = (float*)take(512 * 4);
  float*          scal    = (float*)take(256);

  const float* Ain = x;
  for (int layer = 0; layer < 3; ++layer) {
    float* Hout = out + (size_t)layer * ND;

    // H = A @ W  (bf16 operands, f32 accumulate)
    cvt_kernel<<<8192, 256, 0, stream>>>(Ain, Ab, (int)ND);
    cvt_kernel<<<1024, 256, 0, stream>>>(W[layer], Wb, (int)DD);
    gemm_bf16<<<dim3(512, 8, 1), 256, 0, stream>>>(Ab, 512, Wb, 512, H, 512,
                                                   (unsigned short*)nullptr,
                                                   (const float*)nullptr,
                                                   1.f, 0.f, 512, (size_t)0);

    // mean + center (store Xc in bf16), then transpose for the cov GEMM
    colsum_kernel<<<256, 256, 0, stream>>>(H, colpart);
    meanreduce_kernel<<<2, 256, 0, stream>>>(colpart, meansum);
    center_kernel<<<8192, 256, 0, stream>>>(H, meansum, Xcb, (int)ND);
    transpose_bf16<<<dim3(1024, 8), 256, 0, stream>>>(Xcb, XcT);

    // cov = Xc^T Xc (split-K WMMA GEMM, deterministic partial reduction)
    gemm_bf16<<<dim3(4, 8, 32), 256, 0, stream>>>(XcT, 65536, Xcb, 512,
                                                  covpart, 512,
                                                  (unsigned short*)nullptr,
                                                  (const float*)nullptr,
                                                  1.f, 0.f, 2048, DD);
    covreduce_kernel<<<1024, 256, 0, stream>>>(covpart, covacc);
    trace_kernel<<<1, 256, 0, stream>>>(covacc, scal);
    sconvert_kernel<<<1024, 256, 0, stream>>>(covacc, scal, Sb);

    // Newton-Schulz: P <- 1.5P - 0.5 * P @ P @ (P @ S)
    pinit_kernel<<<1024, 256, 0, stream>>>(P0, Pb0);
    float* P = P0; float* Pn = P1;
    unsigned short* Pb = Pb0; unsigned short* Pbn = Pb1;
    for (int it = 0; it < 10; ++it) {
      gemm_bf16<<<dim3(4, 8, 1), 256, 0, stream>>>(Pb, 512, Sb, 512, T1, 512, T1b,
                                                   (const float*)nullptr,
                                                   1.f, 0.f, 512, (size_t)0);
      gemm_bf16<<<dim3(4, 8, 1), 256, 0, stream>>>(Pb, 512, T1b, 512, T2, 512, T2b,
                                                   (const float*)nullptr,
                                                   1.f, 0.f, 512, (size_t)0);
      gemm_bf16<<<dim3(4, 8, 1), 256, 0, stream>>>(Pb, 512, T2b, 512, Pn, 512, Pbn,
                                                   P, -0.5f, 1.5f, 512, (size_t)0);
      float* tf = P; P = Pn; Pn = tf;
      unsigned short* tb = Pb; Pb = Pbn; Pbn = tb;
    }

    // C = P / sqrt(tr) ; out = Xc @ C
    cvt_scale_kernel<<<1024, 256, 0, stream>>>(P, scal + 1, Cb, (int)DD);
    gemm_bf16<<<dim3(512, 8, 1), 256, 0, stream>>>(Xcb, 512, Cb, 512, Hout, 512,
                                                   (unsigned short*)nullptr,
                                                   (const float*)nullptr,
                                                   1.f, 0.f, 512, (size_t)0);
    Ain = Hout;
  }
}
